// E41DiagonalWxCell_79551384257208
// MI455X (gfx1250) — matrix operand order repaired and checked by reading, not compiled, not run on hardware
//
#include <hip/hip_runtime.h>
#include <hip/hip_bf16.h>

typedef _Float16 v16h __attribute__((ext_vector_type(16)));
typedef float    v8f  __attribute__((ext_vector_type(8)));

#define T_STEPS 2048
#define BATCH   16
#define DIM     1024
#define NWG     16
#define COLS_PER_WG 64
#define THREADS 128           // 4 waves of 32
#define NKB     (DIM / 32)    // 32 K-blocks of 32
#define APIPE   4             // A ds-load rotation depth (tiles)
#define AP_ELEMS (NKB * 2 * 16 * 16)   // 16384 f16 per h-buffer (32 KB)

// ws layout (bytes):
//   [0..3]      scale (float)
//   [4..7]      barrier count (zeroed by spectral_kernel each launch)
//   [8..11]     barrier generation (relative compares only — replay safe)
//   [16384..]   Ap: double-buffered packed h (2 * 16384 f16 = 64 KB)
//   [81920..]   Wp: packed scaled W, f16, DIM*DIM elems (2 MB, L2-resident)
#define WS_AP_OFF 16384
#define WS_WP_OFF 81920

__device__ __forceinline__ float fast_rcp(float x) {
  return __builtin_amdgcn_rcpf(x);
}

__device__ __forceinline__ float fast_tanh(float x) {
#if __has_builtin(__builtin_amdgcn_tanhf)
  return __builtin_amdgcn_tanhf(x);           // V_TANH_F32
#elif __has_builtin(__builtin_amdgcn_tanh_f32)
  return __builtin_amdgcn_tanh_f32(x);        // V_TANH_F32
#else
  // branch-free fallback: tanh(x) = sign(x) * (1 - e)/(1 + e), e = exp(-2|x|)
  float e = __expf(-2.0f * __builtin_fabsf(x));
  float r = (1.0f - e) * fast_rcp(1.0f + e);
  return __builtin_copysignf(r, x);
#endif
}

__device__ __forceinline__ void gbarrier(unsigned* cnt, unsigned* gen) {
  __threadfence();                 // release prior global stores (agent scope)
  __syncthreads();
  if (threadIdx.x == 0) {
    unsigned g = __hip_atomic_load(gen, __ATOMIC_RELAXED, __HIP_MEMORY_SCOPE_AGENT);
    unsigned old = __hip_atomic_fetch_add(cnt, 1u, __ATOMIC_ACQ_REL, __HIP_MEMORY_SCOPE_AGENT);
    if (old == NWG - 1) {
      __hip_atomic_store(cnt, 0u, __ATOMIC_RELAXED, __HIP_MEMORY_SCOPE_AGENT);
      __hip_atomic_fetch_add(gen, 1u, __ATOMIC_ACQ_REL, __HIP_MEMORY_SCOPE_AGENT);
    } else {
      while (__hip_atomic_load(gen, __ATOMIC_ACQUIRE, __HIP_MEMORY_SCOPE_AGENT) == g) {
        __builtin_amdgcn_s_sleep(1);
      }
    }
  }
  __syncthreads();
  __threadfence();                 // acquire: invalidate stale WGP-level cache
}

// ---------------- Kernel 1: spectral normalization (one WG) ----------------
__global__ __launch_bounds__(1024) void spectral_kernel(
    const float* __restrict__ W, const float* __restrict__ u0,
    float* __restrict__ ws_f, unsigned* __restrict__ cnt) {
  __shared__ float sred[1024];
  __shared__ float u_s[1024];
  __shared__ float v_s[1024];
  const int tid = threadIdx.x;
  const float EPS = 1e-8f;

  // u = u0 / ||u0||
  float val = u0[tid];
  sred[tid] = val * val; __syncthreads();
  for (int s = 512; s > 0; s >>= 1) { if (tid < s) sred[tid] += sred[tid + s]; __syncthreads(); }
  float n0 = sqrtf(sred[0]); __syncthreads();
  u_s[tid] = val / n0;
  __syncthreads();

  float normu = 0.0f;
  for (int it = 0; it < 3; ++it) {
    // v = W^T u  (column dot; coalesced across lanes)
    float s = 0.0f;
    for (int i = 0; i < DIM; ++i) s += W[i * DIM + tid] * u_s[i];
    sred[tid] = s * s; __syncthreads();
    for (int r = 512; r > 0; r >>= 1) { if (tid < r) sred[tid] += sred[tid + r]; __syncthreads(); }
    float nv = sqrtf(sred[0]); __syncthreads();
    v_s[tid] = s / (nv + EPS);
    __syncthreads();
    // u = W v  (row dot)
    float s2 = 0.0f;
    for (int j = 0; j < DIM; ++j) s2 += W[tid * DIM + j] * v_s[j];
    sred[tid] = s2 * s2; __syncthreads();
    for (int r = 512; r > 0; r >>= 1) { if (tid < r) sred[tid] += sred[tid + r]; __syncthreads(); }
    normu = sqrtf(sred[0]); __syncthreads();
    u_s[tid] = s2 / (normu + EPS);
    __syncthreads();
  }
  if (tid == 0) {
    // sigma = u . (W v) = ||Wv||^2 / (||Wv|| + EPS)
    float sigma = normu * normu / (normu + EPS);
    ws_f[0] = 0.99f / (sigma + EPS);
    __hip_atomic_store(cnt, 0u, __ATOMIC_RELAXED, __HIP_MEMORY_SCOPE_AGENT);
  }
}

// ---------------- Kernel 2: pack scaled W into WMMA B-operand layout -------
// Dense 32x16 B: lane-half h holds K = 16*h + t, K ascending two-per-VGPR.
// Wp[((kb*2+half)*DIM + n)*16 + t] = f16( W[n][kb*32 + half*16 + t] * scale )
__global__ __launch_bounds__(256) void pack_w_kernel(
    const float* __restrict__ W, const float* __restrict__ ws_f,
    _Float16* __restrict__ Wp) {
  const float scale = ws_f[0];
  for (int e = blockIdx.x * blockDim.x + threadIdx.x; e < DIM * DIM;
       e += gridDim.x * blockDim.x) {
    int t = e & 15;
    int rest = e >> 4;
    int n = rest & (DIM - 1);
    int rest2 = rest >> 10;
    int half = rest2 & 1;
    int kb = rest2 >> 1;
    int k = kb * 32 + half * 16 + t;       // B-layout K map (contiguous halves)
    Wp[e] = (_Float16)(W[n * DIM + k] * scale);
  }
}

// ---------------- Kernel 3: persistent sequential scan ---------------------
__global__ __launch_bounds__(THREADS) void rnn_scan_kernel(
    const float* __restrict__ x, const float* __restrict__ h0,
    const float* __restrict__ dx, const float* __restrict__ bvec,
    float* __restrict__ out, const _Float16* __restrict__ Wp,
    _Float16* __restrict__ ap, unsigned* __restrict__ cnt,
    unsigned* __restrict__ gen) {
  __shared__ _Float16 As[AP_ELEMS];   // 32 KB LDS stage of current packed h
  const int tid = threadIdx.x;
  const int wg = blockIdx.x;
  const int wave = tid >> 5;
  const int lid = tid & 31;
  const int lane16 = lid & 15;
  const int half = lid >> 4;

  const size_t OUTS_SZ = (size_t)T_STEPS * BATCH * DIM;
  float* hsec = out + OUTS_SZ;   // h section: [T+1, B, D]

  // ---- init: zero h-buffer 0 for our k-range; copy h0 row into output ----
  {
    int base = (wg * 2) * 2 * 16 * 16;        // our 64 k-cols = 2 kb blocks
    for (int e = tid; e < 2 * 2 * 16 * 16; e += THREADS)
      ap[base + e] = (_Float16)0.0f;
    for (int e = tid; e < BATCH * COLS_PER_WG; e += THREADS) {
      int m = e / COLS_PER_WG, nl = e % COLS_PER_WG;
      int n = wg * COLS_PER_WG + nl;
      hsec[m * DIM + n] = h0[m * DIM + n];
    }
  }
  gbarrier(cnt, gen);

  // ---- per-lane constants ----
  const int n_lane = wg * COLS_PER_WG + wave * 16 + lane16;  // output column
  const float dxv = dx[n_lane];
  const float bv  = bvec[n_lane];
  const int m_base = half * 8;                               // C/D row group

  // inverse A-kmap for scattering h (k = n_lane) into the A-packed buffer
  // A layout: lane-half 0 -> K{0..15}, lane-half 1 -> K{16..31}
  const int kb2 = n_lane >> 5;
  const int kk  = n_lane & 31;
  int half2, t2;
  half2 = kk >> 4; t2 = kk & 15;
  const int ap_wr = ((kb2 * 2 + half2) * 16 + m_base) * 16 + t2;

  // base pointers: all hot-loop addressing becomes base + immediate offset
  const int a_off = (half * 16 + lane16) * 16;               // A lane offset
  const uint4* Ag0 = (const uint4*)ap;                       // global h bufs
  const uint4* Ag1 = (const uint4*)(ap + AP_ELEMS);
  _Float16* Aw0 = ap + ap_wr;                                // + r*16
  _Float16* Aw1 = ap + AP_ELEMS + ap_wr;
  const _Float16* Bb = Wp + ((size_t)half * DIM + n_lane) * 16;  // + kb*32768

  const float* xt_lane  = x    + (size_t)m_base * DIM + n_lane;  // + r*DIM
  float*       out_lane = out  + (size_t)m_base * DIM + n_lane;
  float*       h_lane   = hsec + (size_t)BATCH * DIM             // h[1]
                               + (size_t)m_base * DIM + n_lane;

  // ---- W tiles live in registers for the whole scan (~256 VGPRs) ----
  v16h Breg[NKB];
#pragma unroll
  for (int kb = 0; kb < NKB; ++kb)
    Breg[kb] = *(const v16h*)(Bb + (size_t)kb * 32768);

  for (int t = 0; t < T_STEPS; ++t) {
    _Float16* Aw = (t & 1) ? Aw0 : Aw1;         // write buffer (t+1)&1

    // ---- stage packed h (buffer t&1) global -> LDS, one burst per WG ----
    {
      const uint4* src = (t & 1) ? Ag1 : Ag0;
      uint4 tmp[16];                            // 2048 chunks / 128 threads
#pragma unroll
      for (int i = 0; i < 16; ++i) tmp[i] = src[tid + i * THREADS];
      uint4* dst = (uint4*)As;
#pragma unroll
      for (int i = 0; i < 16; ++i) dst[tid + i * THREADS] = tmp[i];
    }

    // this step's input slice (consumed after the matmul; overlaps WMMA)
    float xr[8];
#pragma unroll
    for (int r = 0; r < 8; ++r) xr[r] = xt_lane[r * DIM];
    if (t + 1 < T_STEPS) {
#pragma unroll
      for (int r = 0; r < 8; ++r)
        __builtin_prefetch(xt_lane + (size_t)BATCH * DIM + r * DIM, 0, 1);
    }

    __syncthreads();   // LDS stage visible to all 4 waves

    // ---- C = h_prev @ Wn^T : K=1024, 32 WMMA, A tiles from LDS ----
    v16h a[APIPE];
#pragma unroll
    for (int i = 0; i < APIPE; ++i)
      a[i] = *(const v16h*)&As[a_off + i * 512];

    v8f c0 = {}; v8f c1 = {};
#pragma unroll
    for (int kb = 0; kb < NKB; ++kb) {
      if (kb & 1)
        c1 = __builtin_amdgcn_wmma_f32_16x16x32_f16(false, a[kb % APIPE],
                 false, Breg[kb], (short)0, c1, false, false);
      else
        c0 = __builtin_amdgcn_wmma_f32_16x16x32_f16(false, a[kb % APIPE],
                 false, Breg[kb], (short)0, c0, false, false);
      if (kb + APIPE < NKB)
        a[kb % APIPE] = *(const v16h*)&As[a_off + (kb + APIPE) * 512];
    }

    // ---- elementwise: tanh, self-gate, write outputs, repack h ----
#pragma unroll
    for (int r = 0; r < 8; ++r) {
      float lin = c0[r] + c1[r] + dxv * xr[r] + bv;
      float h = fast_tanh(lin);
      float sg = fast_rcp(1.0f + __expf(-h));
      out_lane[r * DIM] = h * h * sg;        // h * silu(h)
      h_lane[r * DIM]   = h;
      Aw[r * 16] = (_Float16)h;              // A-layout pack for next step
    }
    xt_lane  += (size_t)BATCH * DIM;
    out_lane += (size_t)BATCH * DIM;
    h_lane   += (size_t)BATCH * DIM;

    gbarrier(cnt, gen);  // publish h_t; also guards LDS reuse (syncthreads)
  }
}

extern "C" void kernel_launch(void* const* d_in, const int* in_sizes, int n_in,
                              void* d_out, int out_size, void* d_ws, size_t ws_size,
                              hipStream_t stream) {
  const float* x  = (const float*)d_in[0];
  const float* h0 = (const float*)d_in[1];
  const float* dx = (const float*)d_in[2];
  const float* W  = (const float*)d_in[3];
  const float* b  = (const float*)d_in[4];
  const float* u0 = (const float*)d_in[5];
  float* out = (float*)d_out;

  char* ws = (char*)d_ws;
  float*    ws_f = (float*)ws;
  unsigned* cnt  = (unsigned*)(ws + 4);
  unsigned* gen  = (unsigned*)(ws + 8);
  _Float16* ap   = (_Float16*)(ws + WS_AP_OFF);
  _Float16* Wp   = (_Float16*)(ws + WS_WP_OFF);

  spectral_kernel<<<1, 1024, 0, stream>>>(W, u0, ws_f, cnt);
  pack_w_kernel<<<1024, 256, 0, stream>>>(W, ws_f, Wp);
  rnn_scan_kernel<<<NWG, THREADS, 0, stream>>>(x, h0, dx, b, out, Wp, ap, cnt, gen);
}
